// Deform_DFN_42442866819728
// MI455X (gfx1250) — matrix operand order repaired and checked by reading, not compiled, not run on hardware
//
#include <hip/hip_runtime.h>
#include <cstddef>
#include <cstdint>

typedef _Float16 h16;
typedef _Float16 v8h  __attribute__((ext_vector_type(8)));
typedef _Float16 v16h __attribute__((ext_vector_type(16)));
typedef float    v8f  __attribute__((ext_vector_type(8)));

#define USE_ASYNC_LDS 1
#define LDSS 72   // LDS row stride in halves (144B = 9*16B: 16B-aligned fragments)

// ---------------------------------------------------------------------------
// Implicit-GEMM conv: out[n][m] = sum_k A[m][k] * W[n][k] + bias[n], m=(b,ho,wo)
// A built on the fly (im2col) into LDS, W pre-converted fp16 [Npad][Kpad64].
// Fuses bias + LeakyReLU + per-channel sum/sumsq (BatchNorm stats) epilogue.
// Tile: 128(M) x 64(N), 256 threads = 8 wave32s; K-step 64 -> 8 WMMAs/wave/iter.
// Weight tile streamed global->LDS with async-to-LDS (ASYNCcnt) on CDNA5.
// ---------------------------------------------------------------------------
__global__ __launch_bounds__(256)
void conv_wmma_kernel(const float* __restrict__ act_in,
                      const h16*  __restrict__ w16,
                      const float* __restrict__ bias,
                      float* __restrict__ act_out,
                      float* __restrict__ stats,     // [2*Nreal] or nullptr
                      int Cin, int Hin, int Win,
                      int Hout, int Wout,
                      int ksz, int stride, int pad,
                      int Kpad, int Nreal, int lrelu)
{
    __shared__ h16 As[128][LDSS];
    __shared__ h16 Bs[64][LDSS];

    const int tid  = threadIdx.x;
    const int lane = tid & 31;
    const int wave = tid >> 5;
    const int m0   = blockIdx.x * 128;
    const int n0   = blockIdx.y * 64;
    const int HW   = Hout * Wout;
    const int M    = 16 * HW;

    // im2col coordinates for this thread's A-tile row (computed once)
    const int arow  = tid & 127;
    const int ahalf = tid >> 7;          // 0/1 -> k-columns [0,32) / [32,64)
    const int m  = m0 + arow;
    const int b  = m / HW;
    const int hw = m - b * HW;
    const int ho = hw / Wout;
    const int wo = hw - ho * Wout;
    const int hob = ho * stride - pad;
    const int wob = wo * stride - pad;
    const int plane16 = 16 * Hin * Win;
    const int bplane  = b * Hin * Win;

    v8f acc[4] = {};

    for (int k0 = 0; k0 < Kpad; k0 += 64) {
        // ---- A tile: implicit im2col, 32 halves per thread, no per-elem divs ----
        {
            const int kbase = k0 + ahalf * 32;
            int ci0 = kbase, t0 = 0;
            if (ksz == 3) { ci0 = kbase / 9; t0 = kbase - ci0 * 9; }
            #pragma unroll
            for (int e = 0; e < 32; ++e) {
                int ci, hi, wi;
                if (ksz == 3) {
                    int t = t0 + e;                       // 0..39
                    int d = (t >= 9) + (t >= 18) + (t >= 27) + (t >= 36);
                    ci = ci0 + d;
                    int tt = t - 9 * d;                   // 0..8
                    int kh = (tt >= 3) + (tt >= 6);
                    int kw = tt - 3 * kh;
                    hi = hob + kh; wi = wob + kw;
                } else {
                    ci = kbase + e; hi = hob; wi = wob;
                }
                float v = 0.f;
                if (ci < Cin && (unsigned)hi < (unsigned)Hin && (unsigned)wi < (unsigned)Win)
                    v = act_in[(size_t)ci * plane16 + bplane + hi * Win + wi];
                As[arow][ahalf * 32 + e] = (h16)v;
            }
        }
        // ---- B tile: 64x64 halves, one 16B global->LDS transfer per thread ----
        {
            const int n  = tid & 63;
            const int c0 = (tid >> 6) * 16;               // 0,16,32,48
            const h16* gsrc = &w16[(size_t)(n0 + n) * Kpad + k0 + c0];
#if USE_ASYNC_LDS
            unsigned ldsa = (unsigned)(uintptr_t)&Bs[n][c0];
            asm volatile("global_load_async_to_lds_b128 %0, %1, off"
                         :: "v"(ldsa), "v"(gsrc) : "memory");
#else
            *(v8h*)&Bs[n][c0] = *(const v8h*)gsrc;
#endif
        }
#if USE_ASYNC_LDS
        asm volatile("s_wait_asynccnt 0" ::: "memory");
#endif
        __syncthreads();

        // ---- WMMA: 2 K-substeps x 4 N-blocks per wave ----
        {
            const int ar  = wave * 16 + (lane & 15);
            const int akb = (lane < 16) ? 0 : 8;    // A frag per ISA 16-bit 16x32 layout
            const int bkb = (lane < 16) ? 0 : 8;    // B frag same k order as A (16-bit 16x32 layout)
            #pragma unroll
            for (int s = 0; s < 2; ++s) {
                v8h alo = *(const v8h*)&As[ar][s * 32 + akb];
                v8h ahi = *(const v8h*)&As[ar][s * 32 + akb + 16];
                v16h afrag = __builtin_shufflevector(alo, ahi,
                    0,1,2,3,4,5,6,7,8,9,10,11,12,13,14,15);
                #pragma unroll
                for (int nb = 0; nb < 4; ++nb) {
                    const int bn = nb * 16 + (lane & 15);
                    v8h blo = *(const v8h*)&Bs[bn][s * 32 + bkb];
                    v8h bhi = *(const v8h*)&Bs[bn][s * 32 + bkb + 16];
                    v16h bfrag = __builtin_shufflevector(blo, bhi,
                        0,1,2,3,4,5,6,7,8,9,10,11,12,13,14,15);
                    acc[nb] = __builtin_amdgcn_wmma_f32_16x16x32_f16(
                        false, afrag, false, bfrag, (short)0, acc[nb], false, false);
                }
            }
        }
        __syncthreads();
    }

    // ---- epilogue: bias + LeakyReLU + BN stats + store (C/D layout per ISA) ----
    const int mrowbase = m0 + wave * 16 + ((lane < 16) ? 0 : 8);
    #pragma unroll
    for (int nb = 0; nb < 4; ++nb) {
        const int ng = n0 + nb * 16 + (lane & 15);
        if (ng >= Nreal) continue;
        const float bv = bias[ng];
        float s = 0.f, q = 0.f;
        #pragma unroll
        for (int vr = 0; vr < 8; ++vr) {
            float x = acc[nb][vr] + bv;
            if (lrelu) x = (x >= 0.f) ? x : 0.01f * x;
            s += x; q += x * x;
            act_out[(size_t)ng * M + (mrowbase + vr)] = x;
        }
        if (stats) {
            atomicAdd(&stats[ng], s);
            atomicAdd(&stats[Nreal + ng], q);
        }
    }
}

// ---------------------------------------------------------------------------
__global__ void convert_w_kernel(const float* __restrict__ w, h16* __restrict__ out,
                                 int Nreal, int Kreal, int Kpad, int Npad)
{
    int idx = blockIdx.x * 256 + threadIdx.x;
    if (idx >= Npad * Kpad) return;
    int n = idx / Kpad;
    int k = idx - n * Kpad;
    float v = (n < Nreal && k < Kreal) ? w[(size_t)n * Kreal + k] : 0.f;
    out[idx] = (h16)v;
}

__global__ void bn_apply_kernel(float* __restrict__ act, const float* __restrict__ stats,
                                const float* __restrict__ g, const float* __restrict__ be,
                                int C, int M, int mShift, float invM)
{
    size_t idx = (size_t)blockIdx.x * 256 + threadIdx.x;
    if (idx >= (size_t)C * M) return;
    int c = (int)(idx >> mShift);
    float mean = stats[c] * invM;
    float var  = stats[C + c] * invM - mean * mean;
    float sc   = g[c] * rsqrtf(var + 1e-5f);
    act[idx] = (act[idx] - mean) * sc + be[c];
}

__global__ void extract_frame_kernel(const float* __restrict__ in,
                                     float* __restrict__ frame, int step)
{
    int idx = blockIdx.x * 256 + threadIdx.x;   // 65536 = 16*64*64
    frame[idx] = in[(size_t)idx * 10 + step];
}

__global__ void add2_kernel(float* __restrict__ o, const float* __restrict__ a,
                            const float* __restrict__ b, int n)
{
    int i = blockIdx.x * 256 + threadIdx.x;
    if (i < n) o[i] = a[i] + b[i];
}

__global__ void upsample2x_kernel(const float* __restrict__ in, float* __restrict__ out,
                                  int CB)   // C*16 planes: 32x32 -> 64x64 nearest
{
    int idx = blockIdx.x * 256 + threadIdx.x;
    if (idx >= CB * 4096) return;
    int w = idx & 63, h = (idx >> 6) & 63, cb = idx >> 12;
    out[idx] = in[((size_t)cb << 10) + ((h >> 1) << 5) + (w >> 1)];
}

__global__ void softmax81_kernel(float* __restrict__ df)   // in-place over 81 channels
{
    int m = blockIdx.x * 256 + threadIdx.x;   // 65536 positions
    float vmax = -1e30f;
    for (int n = 0; n < 81; ++n) vmax = fmaxf(vmax, df[(size_t)n * 65536 + m]);
    float s = 0.f;
    for (int n = 0; n < 81; ++n) s += __expf(df[(size_t)n * 65536 + m] - vmax);
    float inv = 1.f / s;
    for (int n = 0; n < 81; ++n) {
        size_t o = (size_t)n * 65536 + m;
        df[o] = __expf(df[o] - vmax) * inv;
    }
}

__global__ void deform_sample_kernel(const float* __restrict__ frame,
                                     const float* __restrict__ off,
                                     float* __restrict__ map)
{
    unsigned gid = blockIdx.x * 256 + threadIdx.x;     // 81 * 65536 threads
    if (gid >= 81u * 65536u) return;
    int t = gid >> 16;
    int m = gid & 65535;
    int b = m >> 12;
    int ij = m & 4095;
    int i = ij >> 6, j = ij & 63;
    int u = t / 9, v = t - u * 9;
    float offx = off[(size_t)t * 65536 + m];
    float offy = off[(size_t)(t + 81) * 65536 + m];
    float px = (float)(i - 3 + u) + offx;   // (i+1) + (u-4)
    float py = (float)(j - 3 + v) + offy;
    float fx = floorf(px), fy = floorf(py);
    float qltx = fminf(fmaxf(fx, 0.f), 65.f);
    float qlty = fminf(fmaxf(fy, 0.f), 65.f);
    float qrbx = fminf(fmaxf(fx + 1.f, 0.f), 65.f);
    float qrby = fminf(fmaxf(fy + 1.f, 0.f), 65.f);
    px = fminf(fmaxf(px, 0.f), 65.f);
    py = fminf(fmaxf(py, 0.f), 65.f);
    float glt = (1.f + (qltx - px)) * (1.f + (qlty - py));
    float grb = (1.f - (qrbx - px)) * (1.f - (qrby - py));
    float glb = (1.f + (qltx - px)) * (1.f - (qrby - py));
    float grt = (1.f - (qrbx - px)) * (1.f + (qlty - py));
    int iltx = (int)qltx, ilty = (int)qlty, irbx = (int)qrbx, irby = (int)qrby;
    const float* fb = frame + ((size_t)b << 12);
    auto samp = [&](int r, int c) -> float {
        r -= 1; c -= 1;
        return ((unsigned)r < 64u && (unsigned)c < 64u) ? fb[(r << 6) + c] : 0.f;
    };
    float val = glt * samp(iltx, ilty) + grb * samp(irbx, irby)
              + glb * samp(iltx, irby) + grt * samp(irbx, ilty);
    map[(size_t)b * 331776 + (size_t)(i * 9 + u) * 576 + (j * 9 + v)] = val;
}

__global__ void predict_kernel(const float* __restrict__ map,
                               const float* __restrict__ df,
                               float* __restrict__ pred,
                               float* __restrict__ out_slot)
{
    int gid = blockIdx.x * 256 + threadIdx.x;   // 65536
    int b = gid >> 12;
    int ij = gid & 4095;
    int i = ij >> 6, j = ij & 63;
    const float* mb = map + (size_t)b * 331776;
    float sum = 0.f;
    #pragma unroll
    for (int u = 0; u < 9; ++u) {
        int r = i * 9 - 4 + u;
        if ((unsigned)r >= 576u) continue;
        #pragma unroll
        for (int v = 0; v < 9; ++v) {
            int c = j * 9 - 4 + v;
            float blk = ((unsigned)c < 576u) ? mb[r * 576 + c] : 0.f;
            sum += blk * df[(size_t)(u * 9 + v) * 65536 + gid];
        }
    }
    pred[gid] = sum;
    out_slot[gid] = sum;
}

// ---------------------------------------------------------------------------
extern "C" void kernel_launch(void* const* d_in, const int* in_sizes, int n_in,
                              void* d_out, int out_size, void* d_ws, size_t ws_size,
                              hipStream_t stream)
{
    (void)in_sizes; (void)n_in; (void)out_size; (void)ws_size;
    const float* input_frames = (const float*)d_in[0];

    enum {EN1=0,EN2,EN3,EN5,MID1,MIDH1,MIDH2,
          DE1,DE2,DE3,DE4,DE5,DE1_2,DE2_2,DE3_2,DE4_2,DE5_2,DYF,DYF2,NL};
    static const int CI[NL] = {1,32,32,64, 64,128,128, 128,64,64,64,64, 128,64,64,64,64, 128,128};
    static const int CO[NL] = {32,32,64,64, 128,128,128, 64,64,64,64,128, 64,64,64,64,128, 81,162};
    static const int KS[NL] = {3,3,3,3, 3,3,3, 3,3,3,3,1, 3,3,3,3,1, 1,1};

    struct CD { int w,b,g,be; int Cin,Cout,ks,Kreal,Kpad,Npad; size_t woff; };
    CD D[NL];
    size_t wtot = 0;
    for (int l = 0; l < NL; ++l) {
        D[l].Cin = CI[l]; D[l].Cout = CO[l]; D[l].ks = KS[l];
        D[l].Kreal = CI[l] * KS[l] * KS[l];
        D[l].Kpad  = (D[l].Kreal + 63) & ~63;     // K-step is 64
        D[l].Npad  = (CO[l] + 63) & ~63;
        if (l < 17) { D[l].w = 1 + l*4; D[l].b = D[l].w+1; D[l].g = D[l].w+2; D[l].be = D[l].w+3; }
        else if (l == DYF)  { D[l].w = 69; D[l].b = 70; D[l].g = -1; D[l].be = -1; }
        else                { D[l].w = 71; D[l].b = 72; D[l].g = -1; D[l].be = -1; }
        D[l].woff = wtot;
        wtot += ((size_t)D[l].Npad * D[l].Kpad * 2 + 255) & ~(size_t)255;
    }

    // ---- workspace arena ----
    size_t cur = 0;
    auto arena = [&](size_t bytes) -> char* {
        char* p = (char*)d_ws + cur;
        cur += (bytes + 255) & ~(size_t)255;
        return p;
    };
    char*  wbase = arena(wtot);
    float* stats = (float*)arena(2 * 192 * sizeof(float));
    float* b0    = (float*)arena((size_t)128 * 16384 * 4);   // 8 MB
    float* b1    = (float*)arena((size_t)128 * 16384 * 4);   // 8 MB
    float* b4    = (float*)arena((size_t)128 * 16384 * 4);   // 8 MB
    float* hs    = (float*)arena((size_t)128 * 16384 * 4);   // 8 MB
    float* b2a   = (float*)arena((size_t)64  * 65536 * 4);   // 16 MB
    float* b2b   = (float*)arena((size_t)64  * 65536 * 4);   // 16 MB
    float* b3    = (float*)arena((size_t)128 * 65536 * 4);   // 32 MB (also deform map)
    float* DF    = (float*)arena((size_t)81  * 65536 * 4);   // dyn-filter logits/weights
    float* OFF   = (float*)arena((size_t)162 * 65536 * 4);   // deform offsets
    float* fA    = (float*)arena((size_t)16 * 4096 * 4);
    float* fB    = (float*)arena((size_t)16 * 4096 * 4);

    // ---- convert all weights to padded fp16 [Npad][Kpad] (deterministic, per call) ----
    for (int l = 0; l < NL; ++l) {
        int total = D[l].Npad * D[l].Kpad;
        convert_w_kernel<<<(total + 255) / 256, 256, 0, stream>>>(
            (const float*)d_in[D[l].w], (h16*)(wbase + D[l].woff),
            D[l].Cout, D[l].Kreal, D[l].Kpad, D[l].Npad);
    }

    auto run_conv = [&](int l, const float* in, float* out,
                        int Hin, int Hout, int stride, bool bn) {
        const CD& L = D[l];
        int pad = (L.ks == 3) ? 1 : 0;
        int M = 16 * Hout * Hout;
        if (bn) hipMemsetAsync(stats, 0, 2 * L.Cout * sizeof(float), stream);
        dim3 grid(M / 128, L.Npad / 64);
        conv_wmma_kernel<<<grid, 256, 0, stream>>>(
            in, (const h16*)(wbase + L.woff), (const float*)d_in[L.b],
            out, bn ? stats : nullptr,
            L.Cin, Hin, Hin, Hout, Hout, L.ks, stride, pad,
            L.Kpad, L.Cout, bn ? 1 : 0);
        if (bn) {
            int mShift = (Hout == 64) ? 16 : 14;
            size_t tot = (size_t)L.Cout * M;
            bn_apply_kernel<<<(unsigned)((tot + 255) / 256), 256, 0, stream>>>(
                out, stats, (const float*)d_in[L.g], (const float*)d_in[L.be],
                L.Cout, M, mShift, 1.f / (float)M);
        }
    };

    hipMemsetAsync(hs, 0, (size_t)128 * 16384 * 4, stream);   // hs = zeros
    float* frame = fA;
    float* pred  = fB;

    for (int i = 0; i < 19; ++i) {
        if (i < 10)
            extract_frame_kernel<<<256, 256, 0, stream>>>(input_frames, frame, i);
        // encoder
        run_conv(EN1, frame, b0, 64, 64, 1, true);
        run_conv(EN2, b0,    b1, 64, 32, 2, true);
        run_conv(EN3, b1,    b0, 32, 32, 1, true);
        run_conv(EN5, b0,    b1, 32, 32, 1, true);
        // mid: hs_new = mid1(enc) + mid_h2(mid_h1(hs))
        run_conv(MID1,  b1, b0, 32, 32, 1, true);
        run_conv(MIDH1, hs, b4, 32, 32, 1, true);
        run_conv(MIDH2, b4, b1, 32, 32, 1, true);
        add2_kernel<<<(128 * 16384) / 256, 256, 0, stream>>>(hs, b0, b1, 128 * 16384);

        if (i >= 9) {
            for (int dec = 0; dec < 2; ++dec) {
                int base = (dec == 0) ? DE1 : DE1_2;
                run_conv(base + 0, hs, b0, 32, 32, 1, true);
                run_conv(base + 1, b0, b1, 32, 32, 1, true);
                upsample2x_kernel<<<(64 * 16 * 4096) / 256, 256, 0, stream>>>(b1, b2a, 64 * 16);
                run_conv(base + 2, b2a, b2b, 64, 64, 1, true);
                run_conv(base + 3, b2b, b2a, 64, 64, 1, true);
                run_conv(base + 4, b2a, b3,  64, 64, 1, true);   // 1x1, pad 0
                if (dec == 0) {
                    run_conv(DYF, b3, DF, 64, 64, 1, false);     // no BN/act
                    softmax81_kernel<<<256, 256, 0, stream>>>(DF);
                } else {
                    run_conv(DYF2, b3, OFF, 64, 64, 1, false);
                }
            }
            // deform sample into b3 (decoder2 output already consumed)
            deform_sample_kernel<<<(81u * 65536u + 255u) / 256u, 256, 0, stream>>>(frame, OFF, b3);
            predict_kernel<<<256, 256, 0, stream>>>(
                b3, DF, pred, (float*)d_out + (size_t)(i - 9) * 65536);
            float* t = frame; frame = pred; pred = t;
        }
    }
}
